// FoldNet_31980326486620
// MI455X (gfx1250) — compile-verified, matrix operand order checked by reading, not run on hardware
//
#include <hip/hip_runtime.h>

#define BB 16
#define NN 2048
#define KK 16
#define FF 512
#define MM 2025

typedef __attribute__((ext_vector_type(16))) __bf16 v16bf;
typedef __attribute__((ext_vector_type(8)))  float  v8f;
typedef __attribute__((ext_vector_type(4)))  unsigned int u32x4;

union ABPack { u32x4 u[2]; v16bf v; };

__device__ __forceinline__ unsigned short f2bf(float f) {
    unsigned int u = __float_as_uint(f);
    unsigned int r = u + 0x7FFFu + ((u >> 16) & 1u);
    return (unsigned short)(r >> 16);
}
__device__ __forceinline__ float bf2f(unsigned short h) {
    return __uint_as_float(((unsigned int)h) << 16);
}

// -------------------- KNN: top-16 by negative squared distance --------------------
__global__ __launch_bounds__(256) void knn_kernel(const float* __restrict__ pts,
                                                  int* __restrict__ idx) {
    __shared__ float sx[NN], sy[NN], sz[NN];
    int b = blockIdx.x >> 3;          // 8 chunks of 256 points per batch
    int chunk = blockIdx.x & 7;
    const float* p = pts + (size_t)b * NN * 3;
    for (int i = threadIdx.x; i < NN; i += blockDim.x) {
        sx[i] = p[i * 3 + 0];
        sy[i] = p[i * 3 + 1];
        sz[i] = p[i * 3 + 2];
    }
    __syncthreads();
    int n = chunk * 256 + threadIdx.x;
    float qx = sx[n], qy = sy[n], qz = sz[n];
    float bd[KK];
    int   bi[KK];
#pragma unroll
    for (int j = 0; j < KK; j++) { bd[j] = -3.0e38f; bi[j] = -1; }
    for (int m = 0; m < NN; m++) {
        float dx = sx[m] - qx, dy = sy[m] - qy, dz = sz[m] - qz;
        float nd = -(dx * dx + dy * dy + dz * dz);
        if (nd > bd[KK - 1]) {
            int j = KK - 1;
            while (j > 0 && bd[j - 1] < nd) {  // strict < keeps earlier index on ties (stable)
                bd[j] = bd[j - 1]; bi[j] = bi[j - 1]; j--;
            }
            bd[j] = nd; bi[j] = m;
        }
    }
    int* o = idx + ((size_t)b * NN + n) * KK;
#pragma unroll
    for (int j = 0; j < KK; j++) o[j] = bi[j];
}

// -------------------- Weight transpose + f32 -> bf16, zero-padded --------------------
__global__ __launch_bounds__(256) void wtrans_kernel(const float* __restrict__ W,
                                                     unsigned short* __restrict__ Wt,
                                                     int K, int N, int Kp, int Np) {
    int i = blockIdx.x * blockDim.x + threadIdx.x;
    if (i >= Np * Kp) return;
    int n = i / Kp, k = i - n * Kp;
    float v = (n < N && k < K) ? W[(size_t)k * N + n] : 0.0f;
    Wt[i] = f2bf(v);
}

// -------------------- Build x0 = [pts(3), cov(9)] padded to 32, bf16 --------------------
__global__ __launch_bounds__(256) void build_x0_kernel(const float* __restrict__ pts,
                                                       const int* __restrict__ idx,
                                                       unsigned short* __restrict__ x0) {
    int i = blockIdx.x * blockDim.x + threadIdx.x;  // b*N + n
    if (i >= BB * NN) return;
    int b = i >> 11;
    const int* id = idx + (size_t)i * KK;
    const float* pb = pts + (size_t)b * NN * 3;
    const float* pp = pts + (size_t)i * 3;
    float a0 = pb[id[0] * 3 + 0], a1 = pb[id[0] * 3 + 1], a2 = pb[id[0] * 3 + 2];
    float c0 = pb[id[1] * 3 + 0], c1 = pb[id[1] * 3 + 1], c2 = pb[id[1] * 3 + 2];
    unsigned short* o = x0 + (size_t)i * 32;
    o[0] = f2bf(pp[0]); o[1] = f2bf(pp[1]); o[2] = f2bf(pp[2]);
    o[3]  = f2bf(a0 * c0); o[4]  = f2bf(a0 * c1); o[5]  = f2bf(a0 * c2);
    o[6]  = f2bf(a1 * c0); o[7]  = f2bf(a1 * c1); o[8]  = f2bf(a1 * c2);
    o[9]  = f2bf(a2 * c0); o[10] = f2bf(a2 * c1); o[11] = f2bf(a2 * c2);
#pragma unroll
    for (int k = 12; k < 32; k++) o[k] = 0;
}

// -------------------- WMMA bf16 GEMM: each wave computes a 32x64 macro-tile --------------------
// Per k-step: 4 A loads + 8 B loads feed 8 WMMAs -> 1.5 loads/WMMA; A and B fragments
// each reused across 4 / 2 WMMAs respectively.
__global__ __launch_bounds__(256) void gemm_bf16_kernel(
    const unsigned short* __restrict__ A,     // bf16 [M x lda], lda multiple of 32
    const unsigned short* __restrict__ Wt,    // bf16 [Np x Kp], Np multiple of 64, zero-padded
    const float* __restrict__ bias,           // f32 [N]
    void* __restrict__ outp,
    int M, int N, int Kp, int lda, int ldc, int relu, int outF32)
{
    int gw   = (int)((blockIdx.x * blockDim.x + threadIdx.x) >> 5);
    int lane = (int)(threadIdx.x & 31);
    int strips = (N + 63) >> 6;               // 64-wide output strips
    int tilesM = (M + 31) >> 5;               // 32-row macro-tiles (stores row-masked)
    if (gw >= tilesM * strips) return;        // wave-uniform: EXEC stays all-ones for WMMA
    int tm = gw / strips, tn = gw - tm * strips;
    int half = lane >> 4;
    int mr   = lane & 15;
    const unsigned short* arow0 = A + (size_t)(tm * 32 + mr) * lda + 8 * half;
    const unsigned short* arow1 = arow0 + (size_t)16 * lda;
    const unsigned short* brow  = Wt + (size_t)(tn * 64 + mr) * Kp + 8 * half;
    size_t bstep = (size_t)16 * Kp;           // advance 16 weight rows per subtile
    v8f z = {0.f,0.f,0.f,0.f,0.f,0.f,0.f,0.f};
    v8f c00 = z, c01 = z, c02 = z, c03 = z;
    v8f c10 = z, c11 = z, c12 = z, c13 = z;
    for (int k0 = 0; k0 < Kp; k0 += 32) {
        ABPack a0, a1, b0, b1, b2, b3;
        // 16-bit A/B layout: element e -> K = 16*(e>>3) + 8*half + (e&7)
        a0.u[0] = *(const u32x4*)(arow0 + k0);
        a0.u[1] = *(const u32x4*)(arow0 + k0 + 16);
        a1.u[0] = *(const u32x4*)(arow1 + k0);
        a1.u[1] = *(const u32x4*)(arow1 + k0 + 16);
        b0.u[0] = *(const u32x4*)(brow + k0);
        b0.u[1] = *(const u32x4*)(brow + k0 + 16);
        b1.u[0] = *(const u32x4*)(brow + bstep + k0);
        b1.u[1] = *(const u32x4*)(brow + bstep + k0 + 16);
        b2.u[0] = *(const u32x4*)(brow + 2 * bstep + k0);
        b2.u[1] = *(const u32x4*)(brow + 2 * bstep + k0 + 16);
        b3.u[0] = *(const u32x4*)(brow + 3 * bstep + k0);
        b3.u[1] = *(const u32x4*)(brow + 3 * bstep + k0 + 16);
        c00 = __builtin_amdgcn_wmma_f32_16x16x32_bf16(false, a0.v, false, b0.v, (short)0, c00, false, false);
        c01 = __builtin_amdgcn_wmma_f32_16x16x32_bf16(false, a0.v, false, b1.v, (short)0, c01, false, false);
        c02 = __builtin_amdgcn_wmma_f32_16x16x32_bf16(false, a0.v, false, b2.v, (short)0, c02, false, false);
        c03 = __builtin_amdgcn_wmma_f32_16x16x32_bf16(false, a0.v, false, b3.v, (short)0, c03, false, false);
        c10 = __builtin_amdgcn_wmma_f32_16x16x32_bf16(false, a1.v, false, b0.v, (short)0, c10, false, false);
        c11 = __builtin_amdgcn_wmma_f32_16x16x32_bf16(false, a1.v, false, b1.v, (short)0, c11, false, false);
        c12 = __builtin_amdgcn_wmma_f32_16x16x32_bf16(false, a1.v, false, b2.v, (short)0, c12, false, false);
        c13 = __builtin_amdgcn_wmma_f32_16x16x32_bf16(false, a1.v, false, b3.v, (short)0, c13, false, false);
    }
    // C/D layout: n = lane&15, m = 8*half + v
    v8f accs[2][4] = {{c00, c01, c02, c03}, {c10, c11, c12, c13}};
#pragma unroll
    for (int s = 0; s < 4; s++) {
        int ocol = tn * 64 + s * 16 + mr;
        if (ocol >= N) continue;              // only trims the N=3 layers
        float bv = bias[ocol];
#pragma unroll
        for (int r = 0; r < 2; r++) {
#pragma unroll
            for (int v = 0; v < 8; v++) {
                int orow = tm * 32 + r * 16 + 8 * half + v;
                if (orow >= M) continue;      // edge macro-tiles (M=16 layers, M=32400 decoder)
                float val = accs[r][s][v] + bv;
                if (relu) val = fmaxf(val, 0.0f);
                if (outF32) ((float*)outp)[(size_t)orow * ldc + ocol] = val;
                else ((unsigned short*)outp)[(size_t)orow * ldc + ocol] = f2bf(val);
            }
        }
    }
}

// -------------------- KNN gather + max over 16 neighbors --------------------
__global__ void gather_max_kernel(const unsigned short* __restrict__ x,
                                  const int* __restrict__ idx,
                                  unsigned short* __restrict__ y, int C) {
    int pt = blockIdx.x;                      // b*N + n
    int b = pt >> 11;
    __shared__ int sid[KK];
    if (threadIdx.x < KK) sid[threadIdx.x] = idx[(size_t)pt * KK + threadIdx.x];
    __syncthreads();
    int c = threadIdx.x;
    const unsigned short* xb = x + (size_t)b * NN * C;
    float m = -3.0e38f;
#pragma unroll
    for (int j = 0; j < KK; j++)
        m = fmaxf(m, bf2f(xb[(size_t)sid[j] * C + c]));
    y[(size_t)pt * C + c] = f2bf(m);
}

// -------------------- Global max over N --------------------
__global__ __launch_bounds__(256) void global_max_kernel(const unsigned short* __restrict__ x,
                                                         unsigned short* __restrict__ y) {
    int i = blockIdx.x * blockDim.x + threadIdx.x;  // b*1024 + c
    if (i >= BB * 1024) return;
    int b = i >> 10, c = i & 1023;
    const unsigned short* xb = x + (size_t)b * NN * 1024 + c;
    float m = -3.0e38f;
    for (int n = 0; n < NN; n++) m = fmaxf(m, bf2f(xb[(size_t)n * 1024]));
    y[i] = f2bf(m);
}

// -------------------- Decoder input builders (Kp = 544) --------------------
__global__ __launch_bounds__(256) void build_h1_kernel(const unsigned short* __restrict__ feat,
                                                       unsigned short* __restrict__ h1) {
    int i = blockIdx.x * blockDim.x + threadIdx.x;
    if (i >= BB * MM * 544) return;
    int row = i / 544, k = i - row * 544;
    int b = row / MM, m = row - b * MM;
    unsigned short v;
    if (k < FF) v = feat[b * FF + k];
    else if (k == FF)     { int gi = m / 45; v = f2bf(-0.3f + (0.6f / 44.0f) * (float)gi); }
    else if (k == FF + 1) { int gj = m % 45; v = f2bf(-0.3f + (0.6f / 44.0f) * (float)gj); }
    else v = 0;
    h1[i] = v;
}

__global__ __launch_bounds__(256) void build_h2_kernel(const unsigned short* __restrict__ feat,
                                                       const unsigned short* __restrict__ fold1,
                                                       unsigned short* __restrict__ h2) {
    int i = blockIdx.x * blockDim.x + threadIdx.x;
    if (i >= BB * MM * 544) return;
    int row = i / 544, k = i - row * 544;
    int b = row / MM;
    unsigned short v;
    if (k < FF) v = feat[b * FF + k];
    else if (k < FF + 3) v = fold1[(size_t)row * 3 + (k - FF)];
    else v = 0;
    h2[i] = v;
}

// -------------------- Host orchestration --------------------
static inline int cdiv(int a, int b) { return (a + b - 1) / b; }

extern "C" void kernel_launch(void* const* d_in, const int* in_sizes, int n_in,
                              void* d_out, int out_size, void* d_ws, size_t ws_size,
                              hipStream_t stream) {
    const float* pts = (const float*)d_in[0];
    const float* W[15]; const float* Bv[15];
    for (int i = 0; i < 15; i++) { W[i] = (const float*)d_in[1 + 2 * i]; Bv[i] = (const float*)d_in[2 + 2 * i]; }
    static const int Kdim[15] = {12, 64, 64, 64, 64, 128, 128, 1024, 512, 514, 512, 512, 515, 512, 512};
    static const int Ndim[15] = {64, 64, 64, 64, 128, 128, 1024, 512, 512, 512, 512, 3, 512, 512, 3};

    char* ws = (char*)d_ws;
    size_t off = 0;
    auto alloc = [&](size_t bytes) -> void* {
        void* p = ws + off;
        off = (off + bytes + 255) & ~(size_t)255;
        return p;
    };

    int*            idx   = (int*)           alloc((size_t)BB * NN * KK * 4);
    unsigned short* x0    = (unsigned short*)alloc((size_t)(BB * NN + 32) * 32 * 2);
    unsigned short* P     = (unsigned short*)alloc((size_t)BB * NN * 1024 * 2 + 65536);  // 64 MB arena (+edge pad)
    unsigned short* Q     = (unsigned short*)alloc((size_t)BB * MM * 544 * 2 + 65536);   // 35 MB arena (+edge pad)
    unsigned short* pool  = (unsigned short*)alloc((size_t)(BB + 32) * 1024 * 2);        // over-alloc: 32-row tiles over-read
    unsigned short* m21   = (unsigned short*)alloc((size_t)(BB + 32) * FF * 2);
    unsigned short* feat  = (unsigned short*)alloc((size_t)BB * FF * 2);
    unsigned short* fold1 = (unsigned short*)alloc((size_t)BB * MM * 3 * 2);

    unsigned short* Wt[15]; int Kp[15], Np[15];
    for (int i = 0; i < 15; i++) {
        Kp[i] = ((Kdim[i] + 31) / 32) * 32;
        Np[i] = ((Ndim[i] + 63) / 64) * 64;   // pad N to 64 so wide strips never mask loads
        Wt[i] = (unsigned short*)alloc((size_t)Np[i] * Kp[i] * 2);
        int tot = Np[i] * Kp[i];
        wtrans_kernel<<<cdiv(tot, 256), 256, 0, stream>>>(W[i], Wt[i], Kdim[i], Ndim[i], Kp[i], Np[i]);
    }

    auto gemm = [&](const unsigned short* A, int wi, void* out, int M, int lda, int ldc,
                    int relu, int of32) {
        int tiles = ((M + 31) >> 5) * ((Ndim[wi] + 63) >> 6);
        gemm_bf16_kernel<<<cdiv(tiles, 8), 256, 0, stream>>>(
            A, Wt[wi], Bv[wi], out, M, Ndim[wi], Kp[wi], lda, ldc, relu, of32);
    };

    const int NP = BB * NN;   // 32768 points
    const int NR = BB * MM;   // 32400 decoder rows

    // ---- Encoder ----
    knn_kernel<<<BB * 8, 256, 0, stream>>>(pts, idx);
    build_x0_kernel<<<cdiv(NP, 256), 256, 0, stream>>>(pts, idx, x0);

    gemm(x0, 0, P, NP, 32, 64, 1, 0);                 // mlp1 layer1 -> P [NP x 64]
    gemm(P, 1, Q, NP, 64, 64, 1, 0);                  // mlp1 layer2 -> Q
    gemm(Q, 2, P, NP, 64, 64, 1, 0);                  // mlp1 layer3 -> P
    gather_max_kernel<<<NP, 64, 0, stream>>>(P, idx, Q, 64);       // local maxpool -> Q
    gemm(Q, 3, P, NP, 64, 64, 0, 0);                  // l1 (linear) -> P
    gemm(P, 4, Q, NP, 64, 128, 1, 0);                 // c1 (relu)   -> Q [NP x 128]
    gather_max_kernel<<<NP, 128, 0, stream>>>(Q, idx, P, 128);     // local maxpool -> P
    gemm(P, 5, Q, NP, 128, 128, 0, 0);                // l2 (linear) -> Q
    gemm(Q, 6, P, NP, 128, 1024, 0, 0);               // c2 (linear) -> P [NP x 1024]
    global_max_kernel<<<cdiv(BB * 1024, 256), 256, 0, stream>>>(P, pool);
    gemm(pool, 7, m21, BB, 1024, 512, 1, 0);          // m2w1 (relu)
    gemm(m21, 8, feat, BB, 512, 512, 0, 0);           // m2w2 -> codeword

    // ---- Decoder fold 1 ----
    build_h1_kernel<<<cdiv(NR * 544, 256), 256, 0, stream>>>(feat, P);
    gemm(P, 9, Q, NR, 544, 512, 1, 0);                // f1w1 (relu)
    gemm(Q, 10, P, NR, 512, 512, 1, 0);               // f1w2 (relu)
    gemm(P, 11, fold1, NR, 512, 3, 0, 0);             // f1w3 -> fold1 [NR x 3]

    // ---- Decoder fold 2 ----
    build_h2_kernel<<<cdiv(NR * 544, 256), 256, 0, stream>>>(feat, fold1, Q);
    gemm(Q, 12, P, NR, 544, 512, 1, 0);               // f2w1 (relu)
    gemm(P, 13, Q, NR, 512, 512, 1, 0);               // f2w2 (relu)
    gemm(Q, 14, d_out, NR, 512, 3, 0, 1);             // f2w3 -> f32 output [B, M, 3]
}
